// Transformer_Enc_88957362635389
// MI455X (gfx1250) — compile-verified, matrix operand order checked by reading, not run on hardware
//
#include <hip/hip_runtime.h>
#include <cstdint>
#include <cstddef>

// ---------------------------------------------------------------------------
// Deformable-DETR encoder forward for MI455X (gfx1250).
// bf16 WMMA GEMMs with LDS-staged B tiles via GLOBAL_LOAD_ASYNC_TO_LDS_B128
// (ASYNCcnt double buffering), fp32 accumulation.
// ---------------------------------------------------------------------------

typedef __bf16 bf16_t;
typedef bf16_t v16bf __attribute__((ext_vector_type(16)));
typedef float  v8f   __attribute__((ext_vector_type(8)));

#define NTOK 12240          // 96*96 + 48*48 + 24*24 + 12*12
#define MTOK 48960          // B * NTOK, B = 4
#define MPAD 49024          // MTOK padded to a multiple of 128 (383 * 128)
#define CDIM 256
#define NHEAD 8
#define HDIM 32
#define DFFD 1024

__device__ __forceinline__ bf16_t f2bf(float f) { return (bf16_t)f; }

__device__ __forceinline__ v8f zero8() {
  v8f z;
#pragma unroll
  for (int i = 0; i < 8; ++i) z[i] = 0.0f;
  return z;
}

// 16-bit A-fragment (16x32) per CDNA5 WMMA layout. Lane L (row = L&15), hi=L>>4:
//   elems 0..7  -> K = k0 + hi*8 + [0..7]
//   elems 8..15 -> K = k0 + 16 + hi*8 + [0..7]
__device__ __forceinline__ v16bf frag_load(const bf16_t* __restrict__ rowp, int k0, int hi) {
  union { uint4 u[2]; v16bf v; } f;
  const bf16_t* p = rowp + k0 + hi * 8;
  f.u[0] = *(const uint4*)(p);
  f.u[1] = *(const uint4*)(p + 16);
  return f.v;
}

// Same fragment sourced from an LDS-staged 32-K chunk: row stride 64 bytes,
// bytes [0..15]=k0..7, [16..31]=k8..15, [32..47]=k16..23, [48..63]=k24..31.
__device__ __forceinline__ v16bf frag_lds(const char* buf, int row, int hi) {
  union { uint4 u[2]; v16bf v; } f;
  const char* p = buf + row * 64 + hi * 16;
  f.u[0] = *(const uint4*)(p);
  f.u[1] = *(const uint4*)(p + 32);
  return f.v;
}

__device__ __forceinline__ float gelu_exact(float x) {
  return 0.5f * x * (1.0f + erff(x * 0.7071067811865475f));
}

// Stage one 32-K chunk of a 128-row B tile (128 x 32 bf16 = 8 KB) into LDS.
// 512 x 16B segments over 256 threads -> 2 async instructions per wave.
__device__ __forceinline__ void stage_b(const bf16_t* __restrict__ Bt, int n0, int K,
                                        int k0, uint32_t ldsbase, int tid) {
#pragma unroll
  for (int j = 0; j < 2; ++j) {
    int seg = tid + j * 256;                 // 0..511
    int row = seg >> 2;                      // 0..127
    int col = seg & 3;                       // 16B segment within 64B row
    const bf16_t* g = Bt + (size_t)(n0 + row) * K + k0 + col * 8;
    uint32_t l = ldsbase + (uint32_t)(row * 64 + col * 16);
    asm volatile("global_load_async_to_lds_b128 %0, %1, off"
                 :: "v"(l), "v"(g) : "memory");
  }
}

// ---------------------------------------------------------------------------
// GEMM: out[M,N] = A[M,K](bf16) * Bt[N,K](bf16)^T + bias, optional GELU.
// Block = 256 threads = 8 waves; block tile 128(M) x 128(N).
// Wave tile = 16(M) x 128(N) = 8 WMMAs per 32-K chunk, A fragment reused x8.
// B tile double-buffered in LDS via async loads (ASYNCcnt); B fragments are
// preloaded into registers so the 8 WMMAs issue back-to-back behind one
// s_wait_dscnt instead of one wait per WMMA.
// Grid = (MPAD/128, Npad/128). Stores guarded by true M (MTOK) and N.
// ---------------------------------------------------------------------------
__global__ __launch_bounds__(256)
void gemm_bf16_wmma(const bf16_t* __restrict__ A, const bf16_t* __restrict__ Bt,
                    const float* __restrict__ bias, float* __restrict__ outF,
                    bf16_t* __restrict__ outB, int N, int K, int act) {
  __shared__ __align__(16) char bbuf[2][128 * 64];   // 2 x 8 KB

  const int tid  = threadIdx.x;
  const int lane = tid & 31;
  const int wave = tid >> 5;
  const int r    = lane & 15;
  const int hi   = lane >> 4;
  const int m0   = (blockIdx.x * 8 + wave) * 16;
  const int n0   = blockIdx.y * 128;

  const uint32_t lds0 = (uint32_t)(size_t)(&bbuf[0][0]);
  const uint32_t lds1 = (uint32_t)(size_t)(&bbuf[1][0]);

  v8f acc[8];
#pragma unroll
  for (int t = 0; t < 8; ++t) acc[t] = zero8();

  const bf16_t* arow = A + (size_t)(m0 + r) * K;
  const int nch = K >> 5;

  // Prologue: stage chunk 0.
  stage_b(Bt, n0, K, 0, lds0, tid);

  for (int c = 0; c < nch; ++c) {
    const int k0 = c << 5;
    if (c + 1 < nch) {
      stage_b(Bt, n0, K, k0 + 32, (c & 1) ? lds0 : lds1, tid);
      asm volatile("s_wait_asynccnt 0x2" ::: "memory");   // current chunk landed
    } else {
      asm volatile("s_wait_asynccnt 0x0" ::: "memory");
    }
    __syncthreads();                                       // all waves' stripes visible

    const char* buf = (c & 1) ? bbuf[1] : bbuf[0];
    v16bf a = frag_load(arow, k0, hi);

    // Preload all B fragments, then stream the WMMAs.
    v16bf bfr[8];
#pragma unroll
    for (int t = 0; t < 8; ++t) bfr[t] = frag_lds(buf, t * 16 + r, hi);
#pragma unroll
    for (int t = 0; t < 8; ++t) {
      acc[t] = __builtin_amdgcn_wmma_f32_16x16x32_bf16(false, a, false, bfr[t],
                                                       (short)0, acc[t], false, false);
    }
    __syncthreads();                                       // done reading before overwrite
  }

  // C/D layout: acc[t][v] = C[m0 + v + 8*hi][n0 + 16*t + r]
#pragma unroll
  for (int t = 0; t < 8; ++t) {
    int n = n0 + t * 16 + r;
    if (n < N) {
      float bia = bias ? bias[n] : 0.0f;
#pragma unroll
      for (int v = 0; v < 8; ++v) {
        int m = m0 + v + 8 * hi;
        if (m < MTOK) {
          float val = acc[t][v] + bia;
          if (act == 1) val = gelu_exact(val);
          if (outF) outF[(size_t)m * N + n] = val;
          if (outB) outB[(size_t)m * N + n] = f2bf(val);
        }
      }
    }
  }
}

// ---------------------------------------------------------------------------
// Weight cast + transpose: W[K,N] fp32 -> Wt[Npad,K] bf16 (rows >= N zeroed).
// ---------------------------------------------------------------------------
__global__ void wcast_t(const float* __restrict__ W, bf16_t* __restrict__ Wt,
                        int K, int N, int Npad) {
  int idx = blockIdx.x * blockDim.x + threadIdx.x;
  if (idx >= Npad * K) return;
  int n = idx / K;
  int k = idx - n * K;
  Wt[idx] = (n < N) ? f2bf(W[(size_t)k * N + n]) : f2bf(0.0f);
}

// ---------------------------------------------------------------------------
// Pack srcs -> x[b,n,c]; poss + level_embed -> pf[b,n,c].
// ---------------------------------------------------------------------------
__global__ void pack_inputs(const float* __restrict__ s0_, const float* __restrict__ s1_,
                            const float* __restrict__ s2_, const float* __restrict__ s3_,
                            const float* __restrict__ p0_, const float* __restrict__ p1_,
                            const float* __restrict__ p2_, const float* __restrict__ p3_,
                            const float* __restrict__ lvl_emb,
                            float* __restrict__ x, float* __restrict__ pf) {
  size_t idx = (size_t)blockIdx.x * blockDim.x + threadIdx.x;
  if (idx >= (size_t)MTOK * CDIM) return;
  int c  = (int)(idx & 255);
  int bn = (int)(idx >> 8);
  int n  = bn % NTOK;
  int b  = bn / NTOK;
  const int ls0[4] = {0, 9216, 11520, 12096};
  const int lhw[4] = {9216, 2304, 576, 144};
  int li = (n < 9216) ? 0 : (n < 11520) ? 1 : (n < 12096) ? 2 : 3;
  int local = n - ls0[li];
  const float* sp; const float* pp;
  switch (li) {
    case 0: sp = s0_; pp = p0_; break;
    case 1: sp = s1_; pp = p1_; break;
    case 2: sp = s2_; pp = p2_; break;
    default: sp = s3_; pp = p3_; break;
  }
  size_t si = ((size_t)b * CDIM + c) * lhw[li] + local;
  x[idx]  = sp[si];
  pf[idx] = pp[si] + lvl_emb[li * CDIM + c];
}

// q = bf16(x + pf), xbf = bf16(x)
__global__ void make_q(const float* __restrict__ x, const float* __restrict__ pf,
                       bf16_t* __restrict__ qbf, bf16_t* __restrict__ xbf) {
  size_t idx = (size_t)blockIdx.x * blockDim.x + threadIdx.x;
  if (idx >= (size_t)MTOK * CDIM) return;
  float xv = x[idx];
  xbf[idx] = f2bf(xv);
  qbf[idx] = f2bf(xv + pf[idx]);
}

// ---------------------------------------------------------------------------
// MS deformable attention sampling. One wave per (b, n, head); lane = channel.
// value [M,256] fp32, off [M,192] fp32, attw [M,96] fp32 -> out bf16 [MPAD,256].
// ---------------------------------------------------------------------------
__global__ __launch_bounds__(256)
void msda_sample(const float* __restrict__ value, const float* __restrict__ offs,
                 const float* __restrict__ attw, bf16_t* __restrict__ out_bf) {
  const int wid  = blockIdx.x * 8 + (threadIdx.x >> 5);
  const int lane = threadIdx.x & 31;          // head-dim channel
  if (wid >= MTOK * NHEAD) return;
  const int h  = wid & 7;
  const int bn = wid >> 3;                    // b*NTOK + n
  const int n  = bn % NTOK;

  const int ls0[4]  = {0, 9216, 11520, 12096};
  const int ldim[4] = {96, 48, 24, 12};
  int li0 = (n < 9216) ? 0 : (n < 11520) ? 1 : (n < 12096) ? 2 : 3;
  int local = n - ls0[li0];
  int w0 = ldim[li0];
  int row = local / w0, col = local - row * w0;
  float refx = (col + 0.5f) / (float)w0;
  float refy = (row + 0.5f) / (float)w0;      // square levels: h == w

  // softmax over NL*NP = 12 attention logits
  const float* ap = attw + (size_t)bn * 96 + h * 12;
  float aval[12]; float amax = -1e30f;
#pragma unroll
  for (int i = 0; i < 12; ++i) { aval[i] = ap[i]; amax = fmaxf(amax, aval[i]); }
  float asum = 0.0f;
#pragma unroll
  for (int i = 0; i < 12; ++i) { aval[i] = __expf(aval[i] - amax); asum += aval[i]; }
  float ainv = 1.0f / asum;

  const float* op = offs + (size_t)bn * 192 + h * 24;  // [NL][NP][2]
  const size_t brow = (size_t)(bn - n);                // b * NTOK
  float out = 0.0f;

#pragma unroll
  for (int li = 0; li < 4; ++li) {
    const int dimv = ldim[li];
    const float* vbase = value + (brow + ls0[li]) * CDIM + h * HDIM + lane;
#pragma unroll
    for (int p = 0; p < 3; ++p) {
      float ofx = op[(li * 3 + p) * 2 + 0];
      float ofy = op[(li * 3 + p) * 2 + 1];
      // loc = ref + off/NP * wh(0.1) * 0.5
      float locx = refx + ofx * (0.1f * 0.5f / 3.0f);
      float locy = refy + ofy * (0.1f * 0.5f / 3.0f);
      float xf = locx * dimv - 0.5f;
      float yf = locy * dimv - 0.5f;
      float x0f = floorf(xf), y0f = floorf(yf);
      float wx = xf - x0f, wy = yf - y0f;
      int x0 = (int)x0f, y0 = (int)y0f;
      float aw = aval[li * 3 + p] * ainv;
      float s = 0.0f;
#pragma unroll
      for (int cy = 0; cy < 2; ++cy) {
        int yi = y0 + cy;
        if (yi < 0 || yi >= dimv) continue;
        float wyc = cy ? wy : (1.0f - wy);
#pragma unroll
        for (int cx = 0; cx < 2; ++cx) {
          int xi = x0 + cx;
          if (xi < 0 || xi >= dimv) continue;
          float wxc = cx ? wx : (1.0f - wx);
          s += wyc * wxc * vbase[(size_t)(yi * dimv + xi) * CDIM];
        }
      }
      out += aw * s;
    }
  }
  out_bf[(size_t)bn * CDIM + h * HDIM + lane] = f2bf(out);
}

// ---------------------------------------------------------------------------
// xout = LayerNorm(xin + yin) * g + b; optional bf16 copy. One token / block.
// ---------------------------------------------------------------------------
__global__ __launch_bounds__(256)
void residual_ln(const float* __restrict__ xin, const float* __restrict__ yin,
                 const float* __restrict__ g, const float* __restrict__ beta,
                 float* __restrict__ xout, bf16_t* __restrict__ xbf) {
  __shared__ float red[CDIM];
  const int c = threadIdx.x;
  const size_t base = (size_t)blockIdx.x * CDIM;
  float v = xin[base + c] + (yin ? yin[base + c] : 0.0f);
  red[c] = v; __syncthreads();
  for (int s = 128; s > 0; s >>= 1) { if (c < s) red[c] += red[c + s]; __syncthreads(); }
  float mean = red[0] * (1.0f / CDIM);
  __syncthreads();
  float dv = v - mean;
  red[c] = dv * dv; __syncthreads();
  for (int s = 128; s > 0; s >>= 1) { if (c < s) red[c] += red[c + s]; __syncthreads(); }
  float var = red[0] * (1.0f / CDIM);
  float o = dv * rsqrtf(var + 1e-5f) * g[c] + beta[c];
  xout[base + c] = o;
  if (xbf) xbf[base + c] = f2bf(o);
}

// ---------------------------------------------------------------------------
// Host orchestration.
// ---------------------------------------------------------------------------
extern "C" void kernel_launch(void* const* d_in, const int* in_sizes, int n_in,
                              void* d_out, int out_size, void* d_ws, size_t ws_size,
                              hipStream_t stream) {
  (void)in_sizes; (void)n_in; (void)out_size; (void)ws_size;
  // setup_inputs dict order: (src,mask,pos) x4, then params
  const float* src[4] = {(const float*)d_in[0], (const float*)d_in[3],
                         (const float*)d_in[6], (const float*)d_in[9]};
  const float* pos[4] = {(const float*)d_in[2], (const float*)d_in[5],
                         (const float*)d_in[8], (const float*)d_in[11]};
  const float* lvl    = (const float*)d_in[12];
  const float* W_off  = (const float*)d_in[13]; const float* b_off  = (const float*)d_in[14];
  const float* W_attw = (const float*)d_in[15]; const float* b_attw = (const float*)d_in[16];
  const float* W_val  = (const float*)d_in[17]; const float* b_val  = (const float*)d_in[18];
  const float* W_out  = (const float*)d_in[19]; const float* b_out  = (const float*)d_in[20];
  const float* ln1g   = (const float*)d_in[21]; const float* ln1b   = (const float*)d_in[22];
  const float* W_ff1  = (const float*)d_in[23]; const float* b_ff1  = (const float*)d_in[24];
  const float* W_ff2  = (const float*)d_in[25]; const float* b_ff2  = (const float*)d_in[26];
  const float* ln2g   = (const float*)d_in[27]; const float* ln2b   = (const float*)d_in[28];
  const float* lnfg   = (const float*)d_in[29]; const float* lnfb   = (const float*)d_in[30];

  char* ws = (char*)d_ws;
  size_t woff = 0;
  auto alloc = [&](size_t bytes) -> void* {
    void* p = ws + woff;
    woff = (woff + bytes + 255) & ~(size_t)255;
    return p;
  };
  float*  x      = (float*)alloc((size_t)MTOK * CDIM * 4);
  float*  pf     = (float*)alloc((size_t)MTOK * CDIM * 4);
  float*  valbuf = (float*)alloc((size_t)MTOK * CDIM * 4);
  float*  x2     = valbuf;   // aliased: msda consumes valbuf before x2 is written
  float*  offb   = (float*)alloc((size_t)MTOK * 192 * 4);
  float*  attb   = (float*)alloc((size_t)MTOK * 96 * 4);
  // GEMM A-side (bf16) buffers padded to MPAD rows (pad rows read, never stored)
  bf16_t* xbf    = (bf16_t*)alloc((size_t)MPAD * CDIM * 2);
  bf16_t* qbf    = (bf16_t*)alloc((size_t)MPAD * CDIM * 2);
  bf16_t* msdabf = (bf16_t*)alloc((size_t)MPAD * CDIM * 2);
  bf16_t* hbf    = (bf16_t*)alloc((size_t)MPAD * DFFD * 2);
  bf16_t* wt[3][6];
  for (int l = 0; l < 3; ++l) {
    wt[l][0] = (bf16_t*)alloc((size_t)256 * 256 * 2);   // val
    wt[l][1] = (bf16_t*)alloc((size_t)256 * 256 * 2);   // off  (padded 192->256)
    wt[l][2] = (bf16_t*)alloc((size_t)128 * 256 * 2);   // attw (padded 96->128)
    wt[l][3] = (bf16_t*)alloc((size_t)256 * 256 * 2);   // out
    wt[l][4] = (bf16_t*)alloc((size_t)1024 * 256 * 2);  // ff1
    wt[l][5] = (bf16_t*)alloc((size_t)256 * 1024 * 2);  // ff2
  }

  // Weight prep (fp32 [K,N] -> bf16 [Npad,K]).
  for (int l = 0; l < 3; ++l) {
    auto T = [&](const float* W, bf16_t* Wt, int K, int N, int Npad) {
      int tot = Npad * K;
      wcast_t<<<(tot + 255) / 256, 256, 0, stream>>>(W, Wt, K, N, Npad);
    };
    T(W_val  + (size_t)l * 256 * 256,  wt[l][0], 256,  256,  256);
    T(W_off  + (size_t)l * 256 * 192,  wt[l][1], 256,  192,  256);
    T(W_attw + (size_t)l * 256 * 96,   wt[l][2], 256,   96,  128);
    T(W_out  + (size_t)l * 256 * 256,  wt[l][3], 256,  256,  256);
    T(W_ff1  + (size_t)l * 256 * 1024, wt[l][4], 256, 1024, 1024);
    T(W_ff2  + (size_t)l * 1024 * 256, wt[l][5], 1024, 256,  256);
  }

  // Pack inputs.
  {
    size_t tot = (size_t)MTOK * CDIM;
    pack_inputs<<<(unsigned)((tot + 255) / 256), 256, 0, stream>>>(
        src[0], src[1], src[2], src[3], pos[0], pos[1], pos[2], pos[3], lvl, x, pf);
  }

  auto GEMM = [&](const bf16_t* A, const bf16_t* Bt, const float* bias,
                  float* oF, bf16_t* oB, int N, int Npad, int K, int act) {
    dim3 grid(MPAD / 128, Npad / 128);
    gemm_bf16_wmma<<<grid, 256, 0, stream>>>(A, Bt, bias, oF, oB, N, K, act);
  };

  for (int l = 0; l < 3; ++l) {
    {
      size_t tot = (size_t)MTOK * CDIM;
      make_q<<<(unsigned)((tot + 255) / 256), 256, 0, stream>>>(x, pf, qbf, xbf);
    }
    GEMM(xbf, wt[l][0], b_val  + (size_t)l * 256,  valbuf, nullptr, 256,  256,  256, 0);
    GEMM(qbf, wt[l][1], b_off  + (size_t)l * 192,  offb,   nullptr, 192,  256,  256, 0);
    GEMM(qbf, wt[l][2], b_attw + (size_t)l * 96,   attb,   nullptr,  96,  128,  256, 0);
    msda_sample<<<MTOK, 256, 0, stream>>>(valbuf, offb, attb, msdabf);
    GEMM(msdabf, wt[l][3], b_out + (size_t)l * 256, x2, nullptr, 256, 256, 256, 0);
    residual_ln<<<MTOK, 256, 0, stream>>>(x, x2, ln1g + (size_t)l * 256, ln1b + (size_t)l * 256, x, xbf);
    GEMM(xbf, wt[l][4], b_ff1 + (size_t)l * 1024, nullptr, hbf, 1024, 1024, 256, 1);
    GEMM(hbf, wt[l][5], b_ff2 + (size_t)l * 256,  x2, nullptr, 256, 256, 1024, 0);
    residual_ln<<<MTOK, 256, 0, stream>>>(x, x2, ln2g + (size_t)l * 256, ln2b + (size_t)l * 256, x, xbf);
  }
  // Final LayerNorm directly into d_out (fp32, B*N*C).
  residual_ln<<<MTOK, 256, 0, stream>>>(x, nullptr, lnfg, lnfb, (float*)d_out, nullptr);
}
